// PolarLinearLISTAEncoder_85779086836181
// MI455X (gfx1250) — matrix operand order repaired and checked by reading, not compiled, not run on hardware
//
#include <hip/hip_runtime.h>
#include <hip/hip_bf16.h>

// ---------------------------------------------------------------------------
// PolarLinearLISTAEncoder for MI455X (gfx1250), bf16 WMMA pipeline.
// Compute-bound (~223 GFLOP vs ~300MB HBM traffic), so all eight GEMM stages
// run through one generic v_wmma_f32_16x16x32_bf16 kernel (f32 accumulation)
// with fused fp32 epilogues. A-tiles are staged into LDS with the CDNA5
// async copy path (global_load_async_to_lds_b128 / s_wait_asynccnt);
// B-tiles are transposed into LDS through VGPRs.
// ---------------------------------------------------------------------------

typedef __attribute__((ext_vector_type(16))) __bf16 v16bf;
typedef __attribute__((ext_vector_type(8)))  float  v8f;

#define TB 256
#define BM 128
#define BN 128
#define BK 32

enum { M_PROJ = 0, M_REC = 1, M_RELU = 2, M_SP = 3, M_LIN = 4 };

union FragU { uint4 q[2]; v16bf v; };

__device__ __forceinline__ float softshrink(float z, float th) {
  float a = fabsf(z) - th;
  a = a > 0.f ? a : 0.f;
  return z >= 0.f ? a : -a;       // sign(z)*relu(|z|-th)
}

__device__ __forceinline__ float softplusf(float x) {
  if (x > 20.f)  return x;
  if (x < -20.f) return expf(x);
  return log1pf(expf(x));
}

// ---------------------------------------------------------------------------
// Generic GEMM: out = epilogue(A[MxK] @ W[KxN]).
// A, W are bf16 row-major. M = gridDim.y*BM, N = gridDim.x*BN.
// 8 waves per block arranged 2(M) x 4(N); each wave computes 64x32 via
// 4x2 tiles of v_wmma_f32_16x16x32_bf16. Double-buffered LDS staging:
//   A tile: global -> LDS directly via async copy (ASYNCcnt)
//   B tile: global -> VGPR -> transposed ds stores ([n][k] layout)
// Double-buffer selection is a uniform constant-stride offset (no per-lane
// cndmask chains).
// ---------------------------------------------------------------------------
__global__ __launch_bounds__(TB)
void gemm_bf16_wmma(const __hip_bfloat16* __restrict__ Abuf, int lda,
                    const __hip_bfloat16* __restrict__ Wbuf, int ldw,
                    int K, int N,
                    const float* __restrict__ bias,
                    const float* __restrict__ thr,
                    const float* __restrict__ invs,
                    const float* __restrict__ drv,
                    float* __restrict__ f32o, int f32ld,
                    __hip_bfloat16* __restrict__ bfo, int bfld,
                    __hip_bfloat16* __restrict__ bfo2, int bf2ld,
                    float addc, int mode)
{
  __shared__ __align__(16) __hip_bfloat16 sA[2][BM][BK];      // row-major tile
  __shared__ __align__(16) __hip_bfloat16 sB[2][BN][BK];      // transposed [n][k]

  const int tid  = threadIdx.x;
  const int wave = tid >> 5, lane = tid & 31;
  const int m16  = lane & 15, s = lane >> 4;
  const int wm   = wave >> 2;      // 0..1  -> 64-row strip
  const int wn   = wave & 3;       // 0..3  -> 32-col strip
  const int blockM = blockIdx.y * BM;
  const int blockN = blockIdx.x * BN;

  v8f acc[4][2] = {};

  const int KT = K / BK;           // all K are multiples of 32 by construction

  // Per-thread staging addresses, advanced incrementally (no per-iter mul).
  // Only buffer-0 addresses are kept; buffer 1 is a constant stride away.
  constexpr unsigned A_BUF_BYTES = BM * BK * 2;   // 8192
  constexpr int      B_BUF_ELEMS = BN * BK;       // 4096

  const __hip_bfloat16* pA[2];     // global source, 16B chunk per thread
  unsigned dA0[2];                 // LDS dest byte offset in buffer 0
  #pragma unroll
  for (int c = 0; c < 2; ++c) {
    int lin = tid + c * TB;                  // 512 b128 chunks for 128x32 tile
    int row = lin >> 2, seg = lin & 3;
    pA[c] = Abuf + (size_t)(blockM + row) * lda + seg * 8;
    dA0[c] = (unsigned)(uintptr_t)&sA[0][row][seg * 8];
  }
  const __hip_bfloat16* pB[2];
  __hip_bfloat16* dB0[2];
  #pragma unroll
  for (int c = 0; c < 2; ++c) {
    int lin = tid + c * TB;
    int kr = lin >> 4, seg = lin & 15;       // kr: K row, seg: 8 columns
    pB[c] = Wbuf + (size_t)kr * ldw + blockN + seg * 8;
    dB0[c] = &sB[0][seg * 8][kr];
  }
  const size_t strideB = (size_t)BK * ldw;

  auto stage = [&](int b) {
    // A: async direct-to-LDS copy, tracked by ASYNCcnt.
    const unsigned aoff = b ? A_BUF_BYTES : 0u;   // uniform select
    #pragma unroll
    for (int c = 0; c < 2; ++c) {
      unsigned dst = dA0[c] + aoff;
      asm volatile("global_load_async_to_lds_b128 %0, %1, off"
                   :: "v"(dst), "v"(pA[c]) : "memory");
      pA[c] += BK;
    }
    // B: VGPR bounce + transposed scalar stores into [n][k].
    const int boff = b ? B_BUF_ELEMS : 0;         // uniform select
    #pragma unroll
    for (int c = 0; c < 2; ++c) {
      uint4 d = *(const uint4*)pB[c];
      pB[c] += strideB;
      const __hip_bfloat16* h = (const __hip_bfloat16*)&d;
      __hip_bfloat16* dst = dB0[c] + boff;
      #pragma unroll
      for (int u = 0; u < 8; ++u) dst[u * BK] = h[u];
    }
  };

  stage(0);
  asm volatile("s_wait_asynccnt 0x0" ::: "memory");
  __syncthreads();

  int buf = 0;
  for (int kt = 0; kt < KT; ++kt) {
    const int nb = buf ^ 1;
    if (kt + 1 < KT) stage(nb);

    // ISA 7.12.2 16-bit A layout: lane m(+16s) holds K = {s*8+j, 16+s*8+j}
    FragU af[4], bfr[2];
    #pragma unroll
    for (int tm = 0; tm < 4; ++tm) {
      const uint4* p = (const uint4*)&sA[buf][wm * 64 + tm * 16 + m16][0];
      af[tm].q[0] = p[s];
      af[tm].q[1] = p[2 + s];
    }
    // B staged transposed: lane n(+16s) holds its column's K = {s*8+j, 16+s*8+j}
    #pragma unroll
    for (int tn = 0; tn < 2; ++tn) {
      const uint4* p = (const uint4*)&sB[buf][wn * 32 + tn * 16 + m16][0];
      bfr[tn].q[0] = p[s];
      bfr[tn].q[1] = p[2 + s];
    }
    #pragma unroll
    for (int tm = 0; tm < 4; ++tm)
      #pragma unroll
      for (int tn = 0; tn < 2; ++tn)
        acc[tm][tn] = __builtin_amdgcn_wmma_f32_16x16x32_bf16(
            false, af[tm].v, false, bfr[tn].v, (short)0, acc[tm][tn],
            false, false);

    // Make this wave's async LDS writes for buffer `nb` visible, then
    // rendezvous: after the barrier every wave's writes are complete.
    asm volatile("s_wait_asynccnt 0x0" ::: "memory");
    __syncthreads();
    buf = nb;
  }

  // Epilogue. C/D layout: VGPR r -> M = r + 8*s, N = lane%16.
  #pragma unroll
  for (int tm = 0; tm < 4; ++tm) {
    #pragma unroll
    for (int tn = 0; tn < 2; ++tn) {
      const int gm0 = blockM + wm * 64 + tm * 16 + 8 * s;
      const int gn  = blockN + wn * 32 + tn * 16 + m16;
      #pragma unroll
      for (int r = 0; r < 8; ++r) {
        const int gm = gm0 + r;
        float v = acc[tm][tn][r];
        if (mode == M_PROJ) {
          // drive = inv_s * (x @ Wp) + b  (row-scale folded into epilogue)
          v = v * invs[gm] + bias[gn];
          f32o[(size_t)gm * f32ld + gn] = v;              // drive fp32
          v = softshrink(v, thr[gn]);                     // h0
          bfo[(size_t)gm * bfld + gn] = __float2bfloat16(v);
        } else if (mode == M_REC) {
          v += drv[(size_t)gm * N + gn];
          v = softshrink(v, thr[gn]);
          bfo[(size_t)gm * bfld + gn] = __float2bfloat16(v);
          if (bfo2) bfo2[(size_t)gm * bf2ld + gn] = __float2bfloat16(v);
          if (f32o) f32o[(size_t)gm * f32ld + gn] = v;    // h_delta fp32
        } else if (mode == M_RELU) {
          v += bias[gn];
          v = v > 0.f ? v : 0.f;
          bfo[(size_t)gm * bfld + gn] = __float2bfloat16(v);
          if (f32o) f32o[(size_t)gm * f32ld + gn] = v;    // h_gamma fp32
        } else if (mode == M_SP) {
          v = softplusf(v + bias[gn]) + addc;
          f32o[(size_t)gm * f32ld + gn] = v;
        } else { // M_LIN
          v += bias[gn];
          f32o[(size_t)gm * f32ld + gn] = v;
        }
      }
    }
  }
}

// ---------------------------------------------------------------------------
// Per-row L2 norm + build gamma_in row: [bf16(x) | bf16(log(s+eps)) | h slot | 0 pad]
// ---------------------------------------------------------------------------
__global__ __launch_bounds__(TB)
void norm_kernel(const float* __restrict__ x,
                 float* __restrict__ scale_out,
                 float* __restrict__ invs,
                 __hip_bfloat16* __restrict__ gamma,
                 int D, int GK, int PAD0, int NPAD, float eps)
{
  const int row = blockIdx.x;
  const float* xr = x + (size_t)row * D;
  float ssum = 0.f;
  for (int i = threadIdx.x; i < D; i += TB) { float t = xr[i]; ssum += t * t; }
  __shared__ float red[TB];
  red[threadIdx.x] = ssum;
  __syncthreads();
  for (int off = TB / 2; off > 0; off >>= 1) {
    if (threadIdx.x < off) red[threadIdx.x] += red[threadIdx.x + off];
    __syncthreads();
  }
  const float sc = sqrtf(red[0]);
  __hip_bfloat16* gr = gamma + (size_t)row * GK;
  for (int i = threadIdx.x; i < D; i += TB) gr[i] = __float2bfloat16(xr[i]);
  if (threadIdx.x == 0) {
    scale_out[row] = sc;
    invs[row] = 1.f / (sc + eps);
    gr[D] = __float2bfloat16(logf(sc + eps));   // gain feature
  }
  if (threadIdx.x < NPAD) gr[PAD0 + threadIdx.x] = __float2bfloat16(0.f);
}

__global__ void cvt_f32_bf16(const float* __restrict__ in,
                             __hip_bfloat16* __restrict__ out, int n) {
  int i = blockIdx.x * TB + threadIdx.x;
  if (i < n) out[i] = __float2bfloat16(in[i]);
}

__global__ void zero_bf16(__hip_bfloat16* __restrict__ p, int n) {
  int i = blockIdx.x * TB + threadIdx.x;
  if (i < n) p[i] = __float2bfloat16(0.f);
}

__global__ void exp_f32(const float* __restrict__ in, float* __restrict__ out, int n) {
  int i = blockIdx.x * TB + threadIdx.x;
  if (i < n) out[i] = expf(in[i]);
}

// ---------------------------------------------------------------------------
extern "C" void kernel_launch(void* const* d_in, const int* in_sizes, int n_in,
                              void* d_out, int out_size, void* d_ws, size_t ws_size,
                              hipStream_t stream)
{
  constexpr int Bsz = 16384, Din = 1024, Adim = 512;
  constexpr int GKr = Din + 1 + Adim;    // 1537 real gamma features
  constexpr int GK  = 1568;              // padded to multiple of 32
  constexpr float EPSv = 1e-6f;

  const float* x      = (const float*)d_in[0];
  const float* W_proj = (const float*)d_in[1];
  const float* b_proj = (const float*)d_in[2];
  const float* W_rec  = (const float*)d_in[3];
  const float* logth  = (const float*)d_in[4];
  const float* W_pres = (const float*)d_in[5];
  const float* b_pres = (const float*)d_in[6];
  const float* W_sign = (const float*)d_in[7];
  const float* b_sign = (const float*)d_in[8];
  const float* W_g1   = (const float*)d_in[9];
  const float* b_g1   = (const float*)d_in[10];
  const float* W_g2   = (const float*)d_in[11];
  const float* b_g2   = (const float*)d_in[12];
  const float* W_k    = (const float*)d_in[13];
  const float* b_k    = (const float*)d_in[14];
  const float* W_t    = (const float*)d_in[15];
  const float* b_t    = (const float*)d_in[16];

  float* out = (float*)d_out;
  const size_t BA = (size_t)Bsz * Adim;
  float* o_k     = out;
  float* o_theta = out + BA;
  float* o_pres  = out + 2 * BA;
  float* o_sign  = out + 4 * BA;
  float* o_scale = out + 5 * BA;
  float* o_hdel  = out + 5 * BA + Bsz;
  float* o_hgam  = out + 6 * BA + Bsz;

  // workspace carve-up (256B aligned)
  char* wsB = (char*)d_ws;
  size_t off = 0;
  auto alloc = [&](size_t bytes) -> void* {
    void* p = wsB + off;
    off = (off + bytes + 255) & ~(size_t)255;
    return p;
  };
  auto* wp   = (__hip_bfloat16*)alloc((size_t)Din * Adim * 2);
  auto* wr   = (__hip_bfloat16*)alloc((size_t)Adim * Adim * 2);
  auto* wpr  = (__hip_bfloat16*)alloc((size_t)Adim * 2 * Adim * 2);
  auto* wsg  = (__hip_bfloat16*)alloc((size_t)Adim * Adim * 2);
  auto* wg1  = (__hip_bfloat16*)alloc((size_t)GK * Adim * 2);
  auto* wg2  = (__hip_bfloat16*)alloc((size_t)Adim * Adim * 2);
  auto* wk_  = (__hip_bfloat16*)alloc((size_t)Adim * Adim * 2);
  auto* wt_  = (__hip_bfloat16*)alloc((size_t)Adim * Adim * 2);
  float* thr  = (float*)alloc(Adim * 4);
  float* invs = (float*)alloc((size_t)Bsz * 4);
  auto* gamma = (__hip_bfloat16*)alloc((size_t)Bsz * GK * 2);
  float* drv  = (float*)alloc(BA * 4);
  auto* h0   = (__hip_bfloat16*)alloc(BA * 2);
  auto* h1   = (__hip_bfloat16*)alloc(BA * 2);
  auto* hg1  = (__hip_bfloat16*)alloc(BA * 2);
  auto* hg2  = (__hip_bfloat16*)alloc(BA * 2);

  auto cgrid = [](int n) { return dim3((n + TB - 1) / TB); };

  // --- weight conversion to bf16 (+ zero-padded tail rows of W_g1) ---------
  cvt_f32_bf16<<<cgrid(Din * Adim),      TB, 0, stream>>>(W_proj, wp,  Din * Adim);
  cvt_f32_bf16<<<cgrid(Adim * Adim),     TB, 0, stream>>>(W_rec,  wr,  Adim * Adim);
  cvt_f32_bf16<<<cgrid(Adim * 2 * Adim), TB, 0, stream>>>(W_pres, wpr, Adim * 2 * Adim);
  cvt_f32_bf16<<<cgrid(Adim * Adim),     TB, 0, stream>>>(W_sign, wsg, Adim * Adim);
  cvt_f32_bf16<<<cgrid(GKr * Adim),      TB, 0, stream>>>(W_g1,   wg1, GKr * Adim);
  zero_bf16   <<<cgrid((GK - GKr) * Adim), TB, 0, stream>>>(wg1 + (size_t)GKr * Adim,
                                                            (GK - GKr) * Adim);
  cvt_f32_bf16<<<cgrid(Adim * Adim),     TB, 0, stream>>>(W_g2, wg2, Adim * Adim);
  cvt_f32_bf16<<<cgrid(Adim * Adim),     TB, 0, stream>>>(W_k,  wk_, Adim * Adim);
  cvt_f32_bf16<<<cgrid(Adim * Adim),     TB, 0, stream>>>(W_t,  wt_, Adim * Adim);
  exp_f32     <<<cgrid(Adim),            TB, 0, stream>>>(logth, thr, Adim);

  // --- norms + bf16 x into gamma buffer ------------------------------------
  norm_kernel<<<Bsz, TB, 0, stream>>>(x, o_scale, invs, gamma,
                                      Din, GK, GKr, GK - GKr, EPSv);

  const dim3 blk(TB);
  const dim3 gN512(Adim / BN, Bsz / BM);        // (4, 128)
  const dim3 gN1024(2 * Adim / BN, Bsz / BM);   // (8, 128)

  // --- drive = inv_s * (x @ Wp) + b ; h0 = soft(drive, th) -----------------
  gemm_bf16_wmma<<<gN512, blk, 0, stream>>>(
      gamma, GK, wp, Adim, Din, Adim,
      b_proj, thr, invs, nullptr,
      drv, Adim, h0, Adim, nullptr, 0, 0.f, M_PROJ);

  // --- 15 LISTA iterations: h = soft(drive + h @ W_rec, th) ----------------
  __hip_bfloat16* hc = h0;
  __hip_bfloat16* hn = h1;
  __hip_bfloat16* hdel_gamma = gamma + (Din + 1);    // h columns of gamma_in
  for (int it = 0; it < 15; ++it) {
    const bool last = (it == 14);
    gemm_bf16_wmma<<<gN512, blk, 0, stream>>>(
        hc, Adim, wr, Adim, Adim, Adim,
        nullptr, thr, nullptr, drv,
        last ? o_hdel : nullptr, Adim,
        hn, Adim,
        last ? hdel_gamma : nullptr, GK,
        0.f, M_REC);
    __hip_bfloat16* t = hc; hc = hn; hn = t;
  }
  __hip_bfloat16* hdel = hc;   // aligned bf16 copy of h_delta

  // --- gamma MLP -----------------------------------------------------------
  gemm_bf16_wmma<<<gN512, blk, 0, stream>>>(
      gamma, GK, wg1, Adim, GK, Adim,
      b_g1, nullptr, nullptr, nullptr,
      nullptr, 0, hg1, Adim, nullptr, 0, 0.f, M_RELU);

  gemm_bf16_wmma<<<gN512, blk, 0, stream>>>(
      hg1, Adim, wg2, Adim, Adim, Adim,
      b_g2, nullptr, nullptr, nullptr,
      o_hgam, Adim, hg2, Adim, nullptr, 0, 0.f, M_RELU);

  // --- heads ---------------------------------------------------------------
  gemm_bf16_wmma<<<gN512, blk, 0, stream>>>(
      hg2, Adim, wk_, Adim, Adim, Adim,
      b_k, nullptr, nullptr, nullptr,
      o_k, Adim, nullptr, 0, nullptr, 0, 0.01f, M_SP);

  gemm_bf16_wmma<<<gN512, blk, 0, stream>>>(
      hg2, Adim, wt_, Adim, Adim, Adim,
      b_t, nullptr, nullptr, nullptr,
      o_theta, Adim, nullptr, 0, nullptr, 0, 1e-4f, M_SP);

  gemm_bf16_wmma<<<gN1024, blk, 0, stream>>>(
      hdel, Adim, wpr, 2 * Adim, Adim, 2 * Adim,
      b_pres, nullptr, nullptr, nullptr,
      o_pres, 2 * Adim, nullptr, 0, nullptr, 0, 0.f, M_LIN);

  gemm_bf16_wmma<<<gN512, blk, 0, stream>>>(
      hdel, Adim, wsg, Adim, Adim, Adim,
      b_sign, nullptr, nullptr, nullptr,
      o_sign, Adim, nullptr, 0, nullptr, 0, 0.f, M_LIN);

  (void)in_sizes; (void)n_in; (void)out_size; (void)ws_size;
}
